// SECOND_FSA_65609920414524
// MI455X (gfx1250) — compile-verified, matrix operand order checked by reading, not run on hardware
//
#include <hip/hip_runtime.h>
#include <hip/hip_bf16.h>

// ---------------------------------------------------------------------------
// SECOND + FSA backbone for MI455X (gfx1250, wave32, WMMA + TDM).
// Convs (~200 GFLOP) run as implicit GEMM through v_wmma_f32_16x16x32_f16;
// the weight A-panel is staged into LDS by the Tensor Data Mover per K-chunk.
// ---------------------------------------------------------------------------

typedef __attribute__((ext_vector_type(16))) _Float16 v16h;
typedef __attribute__((ext_vector_type(8)))  _Float16 v8h;
typedef __attribute__((ext_vector_type(8)))  float    v8f;
typedef __attribute__((ext_vector_type(4)))  unsigned int u32x4;
typedef __attribute__((ext_vector_type(4)))  int i32x4;
typedef __attribute__((ext_vector_type(8)))  int i32x8;

#define NPIL 2048
#define BATCH 2
#define CDIM 64
#define NYG 496
#define NXG 432

// ---------------------------------------------------------------------------
// TDM: 2D tile load (global -> LDS).  D# built per CDNA5 ISA §8.3/8.4:
//   group0: count=1 | lds_addr | global_addr | type=2
//   group1: data_size=2B, tensor_dim0/1, tile_dim0/1, tensor_dim0_stride
// Issued by one wave; completion via s_wait_tensorcnt.
// ---------------------------------------------------------------------------
__device__ __forceinline__ void tdm_load_2d_to_lds(unsigned lds_byte_off, const void* gptr,
                                                   unsigned td0, unsigned td1,
                                                   unsigned tile0, unsigned tile1,
                                                   unsigned long stride0_elems)
{
    unsigned long ga = (unsigned long)gptr;
    u32x4 g0;
    g0[0] = 1u;                                                   // count=1, user mode
    g0[1] = lds_byte_off;                                         // lds_addr
    g0[2] = (unsigned)(ga & 0xffffffffu);                         // global_addr[31:0]
    g0[3] = (unsigned)((ga >> 32) & 0x01ffffffu) | (2u << 30);    // global_addr[56:32] | type=2
    i32x8 g1;
    g1[0] = (int)(1u << 16);                                      // data_size = 1 -> 2 bytes
    g1[1] = (int)((td0 & 0xffffu) << 16);                         // tensor_dim0[15:0] @ bit48
    g1[2] = (int)((td0 >> 16) | ((td1 & 0xffffu) << 16));         // td0[31:16] | td1[15:0]
    g1[3] = (int)((td1 >> 16) | (tile0 << 16));                   // td1[31:16] | tile_dim0
    g1[4] = (int)(tile1 & 0xffffu);                               // tile_dim1 (tile_dim2=0)
    g1[5] = (int)(unsigned)(stride0_elems & 0xffffffffu);         // tensor_dim0_stride[31:0]
    g1[6] = (int)(unsigned)((stride0_elems >> 32) & 0xffffu);     // stride0[47:32]
    g1[7] = 0;
    i32x4 z4 = {0, 0, 0, 0};
#if defined(__clang_major__) && (__clang_major__ >= 23)
    i32x8 z8 = {0, 0, 0, 0, 0, 0, 0, 0};
    __builtin_amdgcn_tensor_load_to_lds(g0, g1, z4, z4, z8, 0);
#else
    __builtin_amdgcn_tensor_load_to_lds(g0, g1, z4, z4, 0);
#endif
}

// ---------------------------------------------------------------------------
// f32 -> f16 convert / zero
// ---------------------------------------------------------------------------
__global__ void cvt_f32_f16(const float* __restrict__ s, _Float16* __restrict__ d, int n) {
    int i = blockIdx.x * 256 + threadIdx.x;
    if (i < n) d[i] = (_Float16)s[i];
}

__global__ void zero_f16(_Float16* __restrict__ d, int n) {
    int i = blockIdx.x * 256 + threadIdx.x;
    if (i < n) d[i] = (_Float16)0.0f;
}

// Repack conv weights [Cout][Cin][3][3] (f32) -> [Cout][9][Cin] (f16) so a
// 32-wide K chunk never crosses an (r,s) boundary.
__global__ void repack_w(const float* __restrict__ src, _Float16* __restrict__ dst,
                         int Cout, int Cin) {
    int idx = blockIdx.x * 256 + threadIdx.x;
    int tot = Cout * Cin * 9;
    if (idx >= tot) return;
    int rs = idx % 9; int t = idx / 9;
    int ci = t % Cin; int co = t / Cin;
    dst[((size_t)co * 9 + rs) * Cin + ci] = (_Float16)src[idx];
}

// ---------------------------------------------------------------------------
// Implicit-GEMM 3x3 conv + folded-BN affine + ReLU.
// M = Cout, N = pixels, K = 9*Cin (reordered: k = rs*Cin + cin).
// Block = 256 threads = 8 waves; tile = 128 couts x 32 pixels.
// Each wave: one A fragment (16 couts, from TDM-staged LDS panel) x two B
// fragments (2x16 pixels, im2col-staged LDS) -> 2 wmma per chunk.
// ---------------------------------------------------------------------------
__global__ __launch_bounds__(256)
void conv3x3_bn_relu_wmma(const _Float16* __restrict__ in,   // [B,Cin,IH,IW]
                          const _Float16* __restrict__ w,    // [Cout][9][Cin] f16
                          const float*    __restrict__ gamma,
                          const float*    __restrict__ beta,
                          _Float16*       __restrict__ out,  // [B,Cout,OH,OW]
                          int Cin, int Cout, int IH, int IW, int OH, int OW, int stride)
{
    const int Ktot = Cin * 9;
    const int P    = OH * OW;
    in  += (size_t)blockIdx.z * Cin  * IH * IW;
    out += (size_t)blockIdx.z * Cout * OH * OW;

    const int p0       = blockIdx.x * 32;
    const int coutBase = blockIdx.y * 128;
    const int wave = threadIdx.x >> 5;
    const int lane = threadIdx.x & 31;
    const int hi   = lane >> 4;
    const int lo   = lane & 15;
    const int klo  = hi * 8;

    // LDS: A panel (TDM dest, 128x32 halves, offset 0) + im2col B (32x40)
    __shared__ __align__(16) _Float16 smem[128 * 32 + 32 * 40];
    _Float16* Alds = smem;
    _Float16* Blds = smem + 128 * 32;

    // staging assignment: fixed k-lane (cin offset), 4 fixed pixels per thread
    const int skk = threadIdx.x >> 3;        // 0..31
    const int sn  = (threadIdx.x & 7) * 4;   // 0,4,..,28
    int ohs[4], ows[4]; bool pv[4];
    #pragma unroll
    for (int u = 0; u < 4; ++u) {
        int p = p0 + sn + u;
        pv[u]  = (p < P);
        int oh = p / OW, ow = p % OW;
        ohs[u] = oh * stride - 1;
        ows[u] = ow * stride - 1;
    }

    v8f acc0 = {}, acc1 = {};
    for (int rs = 0; rs < 9; ++rs) {
        const int r = rs / 3, s = rs % 3;
        const size_t wRow = (size_t)coutBase * Ktot + (size_t)rs * Cin;
        for (int ck = 0; ck < Cin; ck += 32) {
            __syncthreads();   // previous chunk fully consumed
            // ---- async: TDM pulls the 128x32 weight panel into LDS ----
            if (threadIdx.x < 32)
                tdm_load_2d_to_lds(0, w + wRow + ck, (unsigned)Ktot, (unsigned)Cout,
                                   32u, 128u, (unsigned long)Ktot);
            // ---- meanwhile all threads stage the im2col B chunk ----
            {
                const _Float16* ip = in + (size_t)(ck + skk) * IH * IW;
                if (ck + 32 < Cin)
                    __builtin_prefetch(ip + 32 * IH * IW, 0, 1);   // global_prefetch_b8
                #pragma unroll
                for (int u = 0; u < 4; ++u) {
                    int ih = ohs[u] + r;
                    int iw = ows[u] + s;
                    _Float16 v = (_Float16)0.0f;
                    if (pv[u] && (unsigned)ih < (unsigned)IH && (unsigned)iw < (unsigned)IW)
                        v = ip[(size_t)ih * IW + iw];
                    Blds[(sn + u) * 40 + skk] = v;
                }
            }
            if (threadIdx.x < 32) __builtin_amdgcn_s_wait_tensorcnt(0);
            __syncthreads();

            // ---- fragments from LDS (ds_load_b128) ----
            const int row = wave * 16 + lo;
            v8h a0  = *(const v8h*)(Alds + row * 32 + klo);
            v8h a1  = *(const v8h*)(Alds + row * 32 + 16 + klo);
            v8h b00 = *(const v8h*)(Blds + lo * 40 + klo);
            v8h b01 = *(const v8h*)(Blds + lo * 40 + 16 + klo);
            v8h b10 = *(const v8h*)(Blds + (16 + lo) * 40 + klo);
            v8h b11 = *(const v8h*)(Blds + (16 + lo) * 40 + 16 + klo);
            v16h a, bA, bB;
            #pragma unroll
            for (int i = 0; i < 8; ++i) {
                a[i] = a0[i];  a[i + 8] = a1[i];
                bA[i] = b00[i]; bA[i + 8] = b01[i];
                bB[i] = b10[i]; bB[i + 8] = b11[i];
            }
            acc0 = __builtin_amdgcn_wmma_f32_16x16x32_f16(false, a, false, bA, (short)0, acc0, false, false);
            acc1 = __builtin_amdgcn_wmma_f32_16x16x32_f16(false, a, false, bB, (short)0, acc1, false, false);
        }
    }

    // ---- epilogue: BN affine + ReLU ----
    const int pA = p0 + lo, pB = p0 + 16 + lo;
    #pragma unroll
    for (int rr = 0; rr < 8; ++rr) {
        int co = coutBase + wave * 16 + rr + hi * 8;
        float g = gamma[co], bb = beta[co];
        if (pA < P) out[(size_t)co * P + pA] = (_Float16)fmaxf(acc0[rr] * g + bb, 0.0f);
        if (pB < P) out[(size_t)co * P + pB] = (_Float16)fmaxf(acc1[rr] * g + bb, 0.0f);
    }
}

// ---------------------------------------------------------------------------
// Generic small GEMM (QKV projections): C[m,n] = sum_k A[m,k]*B[k,n].
// ---------------------------------------------------------------------------
__global__ __launch_bounds__(32)
void gemm_wmma_AB(const _Float16* __restrict__ A, const _Float16* __restrict__ Bm,
                  _Float16* __restrict__ Cm, int M, int K, int N,
                  long bStrideB, long bStrideC)
{
    Bm += (size_t)blockIdx.z * bStrideB;
    Cm += (size_t)blockIdx.z * bStrideC;
    const int lane = threadIdx.x & 31, hi = lane >> 4, lo = lane & 15, klo = hi * 8;
    const int m0 = blockIdx.y * 16, n0 = blockIdx.x * 16;

    v8f acc = {};
    for (int k0 = 0; k0 < K; k0 += 32) {
        v8h a0 = *(const v8h*)(A + (size_t)(m0 + lo) * K + k0 + klo);
        v8h a1 = *(const v8h*)(A + (size_t)(m0 + lo) * K + k0 + 16 + klo);
        v16h a, b;
        #pragma unroll
        for (int i = 0; i < 8; ++i) { a[i] = a0[i]; a[i + 8] = a1[i]; }
        #pragma unroll
        for (int h = 0; h < 16; ++h) {
            int kk = k0 + (h & 7) + ((h >> 3) << 4) + klo;
            b[h] = Bm[(size_t)kk * N + n0 + lo];
        }
        acc = __builtin_amdgcn_wmma_f32_16x16x32_f16(false, a, false, b, (short)0, acc, false, false);
    }
    #pragma unroll
    for (int r = 0; r < 8; ++r)
        Cm[(size_t)(m0 + r + hi * 8) * N + n0 + lo] = (_Float16)acc[r];
}

// ---------------------------------------------------------------------------
// Attention scores: S[n,m] = sum_c q[c,n] k[c,m] per (b,g).  K=16 padded to 32.
// One wave per 16(n) x 64(m) strip: A fragment reused across 4 wmma.
// ---------------------------------------------------------------------------
__global__ __launch_bounds__(32)
void attn_scores_wmma(const _Float16* __restrict__ q, const _Float16* __restrict__ k,
                      float* __restrict__ S, int N)
{
    const int lane = threadIdx.x & 31, hi = lane >> 4, lo = lane & 15, klo = hi * 8;
    const int m0b = blockIdx.x * 64, n0 = blockIdx.y * 16;
    const int bg = blockIdx.z;
    const size_t cbase = ((size_t)(bg >> 2) * CDIM + (size_t)(bg & 3) * 16) * N;
    S += (size_t)bg * N * N;

    v16h a;
    #pragma unroll
    for (int h = 0; h < 16; ++h) {
        int kk = (h & 7) + ((h >> 3) << 4) + klo;
        a[h] = (kk < 16) ? q[cbase + (size_t)kk * N + n0 + lo] : (_Float16)0.0f;
    }
    #pragma unroll
    for (int t = 0; t < 4; ++t) {
        const int m0 = m0b + t * 16;
        v16h b;
        #pragma unroll
        for (int h = 0; h < 16; ++h) {
            int kk = (h & 7) + ((h >> 3) << 4) + klo;
            b[h] = (kk < 16) ? k[cbase + (size_t)kk * N + m0 + lo] : (_Float16)0.0f;
        }
        v8f acc = {};
        acc = __builtin_amdgcn_wmma_f32_16x16x32_f16(false, a, false, b, (short)0, acc, false, false);
        #pragma unroll
        for (int r = 0; r < 8; ++r)
            S[(size_t)(n0 + r + hi * 8) * N + m0 + lo] = acc[r];
    }
}

// ---------------------------------------------------------------------------
// Row softmax, f32 in -> f16 out, one block per row.
// ---------------------------------------------------------------------------
__global__ __launch_bounds__(256)
void softmax_rows(const float* __restrict__ S, _Float16* __restrict__ P, int N)
{
    const size_t rb = ((size_t)blockIdx.y * N + blockIdx.x) * N;
    const float* s = S + rb;
    _Float16*    p = P + rb;
    __shared__ float red[256];
    float mx = -1e30f;
    for (int i = threadIdx.x; i < N; i += 256) mx = fmaxf(mx, s[i]);
    red[threadIdx.x] = mx; __syncthreads();
    for (int o = 128; o > 0; o >>= 1) { if (threadIdx.x < o) red[threadIdx.x] = fmaxf(red[threadIdx.x], red[threadIdx.x + o]); __syncthreads(); }
    mx = red[0]; __syncthreads();
    float sum = 0.0f;
    for (int i = threadIdx.x; i < N; i += 256) sum += __expf(s[i] - mx);
    red[threadIdx.x] = sum; __syncthreads();
    for (int o = 128; o > 0; o >>= 1) { if (threadIdx.x < o) red[threadIdx.x] += red[threadIdx.x + o]; __syncthreads(); }
    float inv = 1.0f / red[0];
    for (int i = threadIdx.x; i < N; i += 256) p[i] = (_Float16)(__expf(s[i] - mx) * inv);
}

// ---------------------------------------------------------------------------
// Attention output: o[c,n] = sum_m t[c,m] P[n,m] per (b,g).
// M=16, N=2048, K=2048; all fragment loads are contiguous b128.
// ---------------------------------------------------------------------------
__global__ __launch_bounds__(32)
void attn_out_wmma(const _Float16* __restrict__ t, const _Float16* __restrict__ P,
                   _Float16* __restrict__ o, int N)
{
    const int lane = threadIdx.x & 31, hi = lane >> 4, lo = lane & 15, klo = hi * 8;
    const int n0 = blockIdx.x * 16;
    const int bg = blockIdx.z;
    const size_t cbase = ((size_t)(bg >> 2) * CDIM + (size_t)(bg & 3) * 16) * N;
    const _Float16* Pg = P + (size_t)bg * N * N;

    v8f acc = {};
    for (int k0 = 0; k0 < N; k0 += 32) {
        v8h a0 = *(const v8h*)(t + cbase + (size_t)lo * N + k0 + klo);
        v8h a1 = *(const v8h*)(t + cbase + (size_t)lo * N + k0 + 16 + klo);
        v8h b0 = *(const v8h*)(Pg + (size_t)(n0 + lo) * N + k0 + klo);
        v8h b1 = *(const v8h*)(Pg + (size_t)(n0 + lo) * N + k0 + 16 + klo);
        v16h a, b;
        #pragma unroll
        for (int i = 0; i < 8; ++i) { a[i] = a0[i]; a[i + 8] = a1[i]; b[i] = b0[i]; b[i + 8] = b1[i]; }
        acc = __builtin_amdgcn_wmma_f32_16x16x32_f16(false, a, false, b, (short)0, acc, false, false);
    }
    #pragma unroll
    for (int r = 0; r < 8; ++r)
        o[cbase + (size_t)(r + hi * 8) * N + n0 + lo] = (_Float16)acc[r];
}

// ---------------------------------------------------------------------------
// Grouped 1x1 conv (wz) + GroupNorm over (cg,N) + affine + residual.
// ---------------------------------------------------------------------------
__global__ __launch_bounds__(256)
void sa_finish(const _Float16* __restrict__ o, const _Float16* __restrict__ wz,
               const float* __restrict__ gng, const float* __restrict__ gnbeta,
               const _Float16* __restrict__ xin, _Float16* __restrict__ xout, int N)
{
    const int bg = blockIdx.x;
    const int g  = bg & 3;
    const size_t cbase = ((size_t)(bg >> 2) * CDIM + (size_t)g * 16) * N;
    const _Float16* wzg = wz + g * 256;
    const int total = 16 * N;

    __shared__ float rs[256], rs2[256];
    float sum = 0.0f, sum2 = 0.0f;
    for (int e = threadIdx.x; e < total; e += 256) {
        int c = e / N, n = e % N;
        float z = 0.0f;
        #pragma unroll
        for (int i = 0; i < 16; ++i)
            z += (float)wzg[c * 16 + i] * (float)o[cbase + (size_t)i * N + n];
        sum += z; sum2 += z * z;
    }
    rs[threadIdx.x] = sum; rs2[threadIdx.x] = sum2; __syncthreads();
    for (int ofs = 128; ofs > 0; ofs >>= 1) {
        if (threadIdx.x < ofs) { rs[threadIdx.x] += rs[threadIdx.x + ofs]; rs2[threadIdx.x] += rs2[threadIdx.x + ofs]; }
        __syncthreads();
    }
    float mu  = rs[0] / (float)total;
    float var = rs2[0] / (float)total - mu * mu;
    float inv = rsqrtf(var + 1e-5f);

    for (int e = threadIdx.x; e < total; e += 256) {
        int c = e / N, n = e % N;
        float z = 0.0f;
        #pragma unroll
        for (int i = 0; i < 16; ++i)
            z += (float)wzg[c * 16 + i] * (float)o[cbase + (size_t)i * N + n];
        int ch = g * 16 + c;
        float v = (z - mu) * inv * gng[ch] + gnbeta[ch] + (float)xin[cbase + (size_t)c * N + n];
        xout[cbase + (size_t)c * N + n] = (_Float16)v;
    }
}

// ---------------------------------------------------------------------------
// Positional encoding + LayerNorm -> [B,C,N] f16.
// ---------------------------------------------------------------------------
__global__ __launch_bounds__(256)
void pos_layernorm(const float* __restrict__ pf, const int* __restrict__ coors,
                   const float* __restrict__ lng, const float* __restrict__ lnb,
                   _Float16* __restrict__ pb, int Mtot)
{
    int i = blockIdx.x * 256 + threadIdx.x;
    if (i >= Mtot) return;
    int bb = coors[i * 4 + 0];
    int y  = coors[i * 4 + 2];
    int x  = coors[i * 4 + 3];
    float v[CDIM];
    const float logk = __logf(10000.0f) / 32.0f;
    #pragma unroll
    for (int j = 0; j < 16; ++j) {
        float dv = __expf(-(float)(2 * j) * logk);
        v[2 * j]          = __sinf((float)x * dv);
        v[2 * j + 1]      = __cosf((float)x * dv);
        v[32 + 2 * j]     = __sinf((float)y * dv);
        v[32 + 2 * j + 1] = __cosf((float)y * dv);
    }
    float mu = 0.0f;
    #pragma unroll
    for (int c = 0; c < CDIM; ++c) { v[c] += pf[(size_t)i * CDIM + c]; mu += v[c]; }
    mu *= (1.0f / CDIM);
    float var = 0.0f;
    #pragma unroll
    for (int c = 0; c < CDIM; ++c) { float d = v[c] - mu; var += d * d; }
    var *= (1.0f / CDIM);
    float inv = rsqrtf(var + 1e-6f);
    int n = i % NPIL;
    #pragma unroll
    for (int c = 0; c < CDIM; ++c)
        pb[((size_t)bb * CDIM + c) * NPIL + n] = (_Float16)((v[c] - mu) * inv * lng[c] + lnb[c]);
}

// ---------------------------------------------------------------------------
// Scatter pillar context into dense BEV (f16).
// ---------------------------------------------------------------------------
__global__ void scatter_ctx(const _Float16* __restrict__ ctx, const int* __restrict__ coors,
                            _Float16* __restrict__ dense, int Mtot)
{
    int idx = blockIdx.x * 256 + threadIdx.x;
    if (idx >= Mtot * CDIM) return;
    int i = idx >> 6, c = idx & 63;
    int bb = coors[i * 4], y = coors[i * 4 + 2], x = coors[i * 4 + 3];
    int n = i % NPIL;
    dense[(((size_t)bb * CDIM + c) * NYG + y) * NXG + x] = ctx[((size_t)bb * CDIM + c) * NPIL + n];
}

// ---------------------------------------------------------------------------
// Bilinear (half-pixel) downsample of dense ctx into the concat output slot.
// ---------------------------------------------------------------------------
__global__ void resize_ctx(const _Float16* __restrict__ dense, float* __restrict__ out,
                           int s, int OHs, int OWs, int CoutF)
{
    int idx = blockIdx.x * 256 + threadIdx.x;
    int tot = BATCH * CDIM * OHs * OWs;
    if (idx >= tot) return;
    int ox = idx % OWs; int tt = idx / OWs;
    int oy = tt % OHs;  tt /= OHs;
    int c  = tt % CDIM; int b = tt / CDIM;
    float sy = ((float)oy + 0.5f) * (float)s - 0.5f;
    float sx = ((float)ox + 0.5f) * (float)s - 0.5f;
    int y0 = (int)floorf(sy), x0 = (int)floorf(sx);
    float fy = sy - (float)y0, fx = sx - (float)x0;
    int y1 = y0 + 1 < NYG ? y0 + 1 : NYG - 1;
    int x1 = x0 + 1 < NXG ? x0 + 1 : NXG - 1;
    y0 = y0 > 0 ? y0 : 0; x0 = x0 > 0 ? x0 : 0;
    const _Float16* dp = dense + ((size_t)b * CDIM + c) * NYG * NXG;
    float v00 = (float)dp[(size_t)y0 * NXG + x0], v01 = (float)dp[(size_t)y0 * NXG + x1];
    float v10 = (float)dp[(size_t)y1 * NXG + x0], v11 = (float)dp[(size_t)y1 * NXG + x1];
    float v = v00 * (1 - fy) * (1 - fx) + v01 * (1 - fy) * fx + v10 * fy * (1 - fx) + v11 * fy * fx;
    out[(((size_t)b * (CoutF + CDIM) + CoutF + c) * OHs + oy) * OWs + ox] = v;
}

// ---------------------------------------------------------------------------
// Copy-convert a stage's final f16 feature map into the concat output (f32).
// ---------------------------------------------------------------------------
__global__ void writeout_feats(const _Float16* __restrict__ f, float* __restrict__ out,
                               int Cf, int H, int W)
{
    int idx = blockIdx.x * 256 + threadIdx.x;
    int tot = BATCH * Cf * H * W;
    if (idx >= tot) return;
    int w = idx % W; int tt = idx / W;
    int h = tt % H;  tt /= H;
    int c = tt % Cf; int b = tt / Cf;
    out[(((size_t)b * (Cf + CDIM) + c) * H + h) * W + w] =
        (float)f[(((size_t)b * Cf + c) * H + h) * W + w];
}

// ---------------------------------------------------------------------------
// Host orchestration
// ---------------------------------------------------------------------------
extern "C" void kernel_launch(void* const* d_in, const int* in_sizes, int n_in,
                              void* d_out, int out_size, void* d_ws, size_t ws_size,
                              hipStream_t stream)
{
    (void)in_sizes; (void)n_in; (void)out_size; (void)ws_size;

    const float* x_f32    = (const float*)d_in[0];
    const float* pillar   = (const float*)d_in[1];
    const int*   coors    = (const int*)d_in[2];
    const float* ln_gamma = (const float*)d_in[51];
    const float* ln_beta  = (const float*)d_in[52];
    const int SA1 = 53, SA2 = 59;   // wt, wp, wg, wz, gn_gamma, gn_beta

    struct CC { int cin, cout, ih, iw, oh, ow, stride; };
    CC cc[16];
    {
        int li = 0;
        cc[li++] = {64, 128, 496, 432, 248, 216, 2};
        for (int j = 0; j < 3; ++j) cc[li++] = {128, 128, 248, 216, 248, 216, 1};
        cc[li++] = {128, 128, 248, 216, 124, 108, 2};
        for (int j = 0; j < 5; ++j) cc[li++] = {128, 128, 124, 108, 124, 108, 1};
        cc[li++] = {128, 256, 124, 108, 62, 54, 2};
        for (int j = 0; j < 5; ++j) cc[li++] = {256, 256, 62, 54, 62, 54, 1};
    }

    char* wsBase = (char*)d_ws;
    size_t wsOff = 0;
    auto alloc = [&](size_t bytes) -> char* {
        char* p = wsBase + wsOff;
        wsOff += (bytes + 255) & ~(size_t)255;
        return p;
    };
    _Float16* w16[16];
    for (int L = 0; L < 16; ++L)
        w16[L] = (_Float16*)alloc((size_t)cc[L].cout * cc[L].cin * 9 * sizeof(_Float16));
    _Float16* saw16[8];
    for (int i = 0; i < 8; ++i)
        saw16[i] = (_Float16*)alloc(((i & 3) == 3 ? 1024 : 4096) * sizeof(_Float16));

    const size_t xElems = (size_t)BATCH * CDIM * NYG * NXG;
    _Float16* x16   = (_Float16*)alloc(xElems * sizeof(_Float16));
    const size_t bigAct = (size_t)BATCH * 128 * 248 * 216;
    _Float16* bufA  = (_Float16*)alloc(bigAct * sizeof(_Float16));
    _Float16* bufB  = (_Float16*)alloc(bigAct * sizeof(_Float16));
    _Float16* feat1 = (_Float16*)alloc(bigAct * sizeof(_Float16));
    _Float16* feat2 = (_Float16*)alloc((size_t)BATCH * 128 * 124 * 108 * sizeof(_Float16));
    _Float16* feat3 = (_Float16*)alloc((size_t)BATCH * 256 * 62 * 54 * sizeof(_Float16));

    const size_t pbElems = (size_t)BATCH * CDIM * NPIL;
    _Float16* pb0 = (_Float16*)alloc(pbElems * sizeof(_Float16));
    _Float16* pb1 = (_Float16*)alloc(pbElems * sizeof(_Float16));
    _Float16* tb  = (_Float16*)alloc(pbElems * sizeof(_Float16));
    _Float16* qb  = (_Float16*)alloc(pbElems * sizeof(_Float16));
    _Float16* kb  = (_Float16*)alloc(pbElems * sizeof(_Float16));
    _Float16* ob  = (_Float16*)alloc(pbElems * sizeof(_Float16));

    const size_t sElems = (size_t)BATCH * 4 * NPIL * NPIL;
    float*    Sbuf = (float*)alloc(sElems * sizeof(float));
    _Float16* Pbuf = (_Float16*)alloc(sElems * sizeof(_Float16));
    _Float16* dense = (_Float16*)alloc(xElems * sizeof(_Float16));

    auto cvt = [&](const float* src, _Float16* dst, size_t n) {
        cvt_f32_f16<<<dim3((unsigned)((n + 255) / 256)), 256, 0, stream>>>(src, dst, (int)n);
    };

    // weight/input conversion (convs use the [Cout][9][Cin] repack)
    cvt(x_f32, x16, xElems);
    for (int L = 0; L < 16; ++L) {
        size_t tot = (size_t)cc[L].cout * cc[L].cin * 9;
        repack_w<<<dim3((unsigned)((tot + 255) / 256)), 256, 0, stream>>>(
            (const float*)d_in[3 + 3 * L], w16[L], cc[L].cout, cc[L].cin);
    }
    for (int s = 0; s < 2; ++s) {
        int base = s ? SA2 : SA1;
        for (int j = 0; j < 4; ++j)
            cvt((const float*)d_in[base + j], saw16[s * 4 + j], j == 3 ? 1024 : 4096);
    }

    // conv pyramid
    {
        const _Float16* src = x16;
        _Float16* chain[2] = {bufA, bufB};
        int pp = 0;
        for (int L = 0; L < 16; ++L) {
            _Float16* dst;
            if (L == 3) dst = feat1;
            else if (L == 9) dst = feat2;
            else if (L == 15) dst = feat3;
            else { dst = chain[pp]; pp ^= 1; }
            const CC& c = cc[L];
            int P = c.oh * c.ow;
            dim3 grid((P + 31) / 32, c.cout / 128, BATCH);
            conv3x3_bn_relu_wmma<<<grid, 256, 0, stream>>>(
                src, w16[L], (const float*)d_in[4 + 3 * L], (const float*)d_in[5 + 3 * L],
                dst, c.cin, c.cout, c.ih, c.iw, c.oh, c.ow, c.stride);
            src = dst;
        }
    }

    // pillar branch
    pos_layernorm<<<dim3(16), 256, 0, stream>>>(pillar, coors, ln_gamma, ln_beta, pb0,
                                                BATCH * NPIL);

    auto run_sa = [&](const _Float16* xin, _Float16* xout, int wBase, int pBase) {
        dim3 gp(NPIL / 16, CDIM / 16, BATCH);
        long bs = (long)CDIM * NPIL;
        gemm_wmma_AB<<<gp, 32, 0, stream>>>(saw16[wBase + 0], xin, tb, CDIM, CDIM, NPIL, bs, bs);
        gemm_wmma_AB<<<gp, 32, 0, stream>>>(saw16[wBase + 1], xin, qb, CDIM, CDIM, NPIL, bs, bs);
        gemm_wmma_AB<<<gp, 32, 0, stream>>>(saw16[wBase + 2], xin, kb, CDIM, CDIM, NPIL, bs, bs);
        attn_scores_wmma<<<dim3(NPIL / 64, NPIL / 16, BATCH * 4), 32, 0, stream>>>(qb, kb, Sbuf, NPIL);
        softmax_rows<<<dim3(NPIL, BATCH * 4), 256, 0, stream>>>(Sbuf, Pbuf, NPIL);
        attn_out_wmma<<<dim3(NPIL / 16, 1, BATCH * 4), 32, 0, stream>>>(tb, Pbuf, ob, NPIL);
        sa_finish<<<dim3(BATCH * 4), 256, 0, stream>>>(
            ob, saw16[wBase + 3], (const float*)d_in[pBase + 4], (const float*)d_in[pBase + 5],
            xin, xout, NPIL);
    };
    run_sa(pb0, pb1, 0, SA1);
    run_sa(pb1, pb0, 4, SA2);

    // dense BEV + outputs
    zero_f16<<<dim3((unsigned)((xElems + 255) / 256)), 256, 0, stream>>>(dense, (int)xElems);
    scatter_ctx<<<dim3((BATCH * NPIL * CDIM + 255) / 256), 256, 0, stream>>>(pb0, coors, dense,
                                                                             BATCH * NPIL);

    float* out0 = (float*)d_out;
    float* out1 = out0 + (size_t)BATCH * 192 * 248 * 216;
    float* out2 = out1 + (size_t)BATCH * 192 * 124 * 108;

    auto wo = [&](const _Float16* f, float* out, int Cf, int H, int W) {
        int tot = BATCH * Cf * H * W;
        writeout_feats<<<dim3((tot + 255) / 256), 256, 0, stream>>>(f, out, Cf, H, W);
    };
    auto rz = [&](float* out, int s, int H, int W, int Cf) {
        int tot = BATCH * CDIM * H * W;
        resize_ctx<<<dim3((tot + 255) / 256), 256, 0, stream>>>(dense, out, s, H, W, Cf);
    };
    wo(feat1, out0, 128, 248, 216);  rz(out0, 2, 248, 216, 128);
    wo(feat2, out1, 128, 124, 108);  rz(out1, 4, 124, 108, 128);
    wo(feat3, out2, 256, 62, 54);    rz(out2, 8, 62, 54, 256);
}
